// SHARNN_35227321762149
// MI455X (gfx1250) — matrix operand order repaired and checked by reading, not compiled, not run on hardware
//
#include <hip/hip_runtime.h>

// ---------------------------------------------------------------------------
// SHA-RNN block for MI455X (gfx1250, wave32, WMMA, 320KB LDS/WGP).
//   1. LN(start) -> bf16
//   2. gates_x = x_ln @ Wih^T + (bih+bhh)        (WMMA bf16 GEMM, 2x2 tiles)
//   3. LSTM recurrence: persistent 32-WG kernel; Whh slice (256KB) LDS-resident,
//      h staged to LDS each step (16-row padded -> guard-free WMMA A-frags),
//      c in LDS, double-buffered h in global, atomic grid barrier per step.
//   4. LN(mem)->k [B,S,E], v [B,E,S] (pre-transposed); LN(mid)->h_mid
//   5. q = qs_g * LN_q(h_mid @ qlW^T + b)
//   6. scores = q @ k^T (batched NT WMMA), softmax(+mask) -> w (bf16)
//   7. mix = w @ v^T-layout (batched NT WMMA, contiguous) -> += h_mid
//   8. out = gelu(LN_xff(h) @ boomW^T + b) + LN_ff(h)
// ---------------------------------------------------------------------------

#define SEQ 2048
#define BB  8
#define EMB 1024
#define G4E 4096
#define SB  (SEQ*BB)   // 16384

typedef __attribute__((ext_vector_type(16))) __bf16 v16bf;
typedef __attribute__((ext_vector_type(8)))  __bf16 v8bf;
typedef __attribute__((ext_vector_type(8)))  float  v8f;

__device__ inline float sigmoidf_(float x) { return 1.0f / (1.0f + __expf(-x)); }

// float -> bf16 bits (round-to-nearest-even)
__device__ inline unsigned short f2bf(float x) {
  unsigned u = __float_as_uint(x);
  unsigned r = u + 0x7fffu + ((u >> 16) & 1u);
  return (unsigned short)(r >> 16);
}

// A-operand (16xK tile of row-major [M,K] bf16), guard-free:
// lanes 0-15: row m0+lane, K = k0+{0..7,16..23}; lanes 16-31: K = k0+{8..15,24..31}.
__device__ inline v16bf load_a_frag(const unsigned short* __restrict__ A,
                                    int lda, int m0, int k0, int lane) {
  int row = m0 + (lane & 15);
  int kb  = k0 + ((lane >> 4) << 3);
  const __bf16* p = (const __bf16*)(A + (size_t)row * lda + kb);
  v8bf lo = *(const v8bf*)p;
  v8bf hi = *(const v8bf*)(p + 16);
  v16bf f;
#pragma unroll
  for (int i = 0; i < 8; ++i) { f[i] = lo[i]; f[8 + i] = hi[i]; }
  return f;
}

// B-operand (Kx16 tile), "NT" layout: B row-major [N,K], contiguous in K.
// lanes 0-15: n = n0+lane, K = k0+{0..7,16..23} ; lanes 16-31: K = k0+{8..15,24..31}.
__device__ inline v16bf load_b_frag(const unsigned short* __restrict__ Bm,
                                    int ldb, int n0, int k0, int lane) {
  int n  = n0 + (lane & 15);
  int kb = k0 + ((lane >> 4) << 3);
  const __bf16* p = (const __bf16*)(Bm + (size_t)n * ldb + kb);
  v8bf lo = *(const v8bf*)p;
  v8bf hi = *(const v8bf*)(p + 16);
  v16bf f;
#pragma unroll
  for (int i = 0; i < 8; ++i) { f[i] = lo[i]; f[8 + i] = hi[i]; }
  return f;
}

__device__ inline v8f wmma_bf16(v16bf a, v16bf b, v8f c) {
  return __builtin_amdgcn_wmma_f32_16x16x32_bf16(false, a, false, b,
                                                 (short)0, c, false, false);
}

__device__ inline v8f zero_acc() {
  v8f a;
#pragma unroll
  for (int i = 0; i < 8; ++i) a[i] = 0.0f;
  return a;
}

__device__ inline void store_tile(float* __restrict__ C, int ldc, int m0, int n,
                                  v8f acc, float bv, int lane) {
  int mb = m0 + ((lane >> 4) << 3);
#pragma unroll
  for (int r = 0; r < 8; ++r) C[(size_t)(mb + r) * ldc + n] = acc[r] + bv;
}

// ---------------------------------------------------------------------------
// Batched GEMM: C[z] = A[z] (MxK bf16 row-major) @ B[z]^T (B: [N,K] bf16) + bias
// Block = 256 thr (8 waves) covers 32(M) x 256(N); each wave: 2x2 16x16 tiles
// (n-tiles at wave*16 and wave*16+128) -> 4 WMMAs per 8 b128 loads.
// Requires M%32==0, N%256==0, K%32==0 (true for all call sites).
// ---------------------------------------------------------------------------
__global__ void __launch_bounds__(256) gemm_bf16_kernel(
    const unsigned short* __restrict__ A, long long aB, int lda,
    const unsigned short* __restrict__ Bm, long long bB, int ldb,
    float* __restrict__ C, long long cB, int ldc,
    const float* __restrict__ bias, int M, int N, int K) {
  const unsigned short* Ab = A  + (size_t)blockIdx.z * aB;
  const unsigned short* Bb = Bm + (size_t)blockIdx.z * bB;
  float*                Cb = C  + (size_t)blockIdx.z * cB;
  int lane = threadIdx.x & 31;
  int wave = threadIdx.x >> 5;
  int m0 = blockIdx.y << 5;
  int n0 = (blockIdx.x << 8) + (wave << 4);

  v8f acc00 = zero_acc(), acc01 = zero_acc();
  v8f acc10 = zero_acc(), acc11 = zero_acc();

  for (int k0 = 0; k0 < K; k0 += 32) {
    if (k0 + 32 < K) {  // -> global_prefetch_b8
      __builtin_prefetch(Ab + (size_t)(m0 + (lane & 15)) * lda + k0 + 32, 0, 1);
      __builtin_prefetch(Bb + (size_t)(n0 + (lane & 15)) * ldb + k0 + 32, 0, 1);
    }
    v16bf a0 = load_a_frag(Ab, lda, m0,      k0, lane);
    v16bf a1 = load_a_frag(Ab, lda, m0 + 16, k0, lane);
    v16bf b0 = load_b_frag(Bb, ldb, n0,       k0, lane);
    v16bf b1 = load_b_frag(Bb, ldb, n0 + 128, k0, lane);
    acc00 = wmma_bf16(a0, b0, acc00);
    acc01 = wmma_bf16(a0, b1, acc01);
    acc10 = wmma_bf16(a1, b0, acc10);
    acc11 = wmma_bf16(a1, b1, acc11);
  }

  int n  = n0 + (lane & 15);
  float bv0 = bias ? bias[n]       : 0.0f;
  float bv1 = bias ? bias[n + 128] : 0.0f;
  store_tile(Cb, ldc, m0,      n,       acc00, bv0, lane);
  store_tile(Cb, ldc, m0,      n + 128, acc01, bv1, lane);
  store_tile(Cb, ldc, m0 + 16, n,       acc10, bv0, lane);
  store_tile(Cb, ldc, m0 + 16, n + 128, acc11, bv1, lane);
}

// ---------------------------------------------------------------------------
// Persistent LSTM recurrence. 32 blocks x 256 thr. Block owns 32 E-columns.
// LDS: Whh slice [128 rows x 1024] bf16 (256KB, loaded once) + padded h
// [16 x 1024] bf16 (32KB, staged per step; rows 8..15 zeroed once).
// Wave w: gate = w>>1 (i,f,g,o), 16 cols each. Atomic grid barrier per step.
// ---------------------------------------------------------------------------
#define WHH_S_ELEMS (128 * EMB)
#define H_S_ELEMS   (16 * EMB)

__global__ void __launch_bounds__(256, 1) lstm_kernel(
    const float* __restrict__ gates_x,        // [S*B, 4E]
    const unsigned short* __restrict__ Whh,   // [4E, E] bf16
    unsigned short* __restrict__ hbuf,        // [2][8][E] bf16 (buf0 zeroed)
    float* __restrict__ rnn_out,              // [S,B,E]
    int* __restrict__ syncv) {
  extern __shared__ unsigned short smem[];
  unsigned short* whh_s = smem;                // [128][1024]
  unsigned short* h_s   = smem + WHH_S_ELEMS;  // [16][1024]
  __shared__ float lds_g[4][8][32];
  __shared__ float lds_c[8][32];

  const int lane = threadIdx.x & 31;
  const int wave = threadIdx.x >> 5;
  const int ec0  = blockIdx.x << 5;
  const int gate = wave >> 1;
  const int lrow0 = (gate << 5) + ((wave & 1) << 4);  // local Whh row base
  const int nblocks = gridDim.x;

  // preload this block's Whh slice: local row lrow = g*32 + c  <-  Whh row g*1024+ec0+c
  for (int i = threadIdx.x; i < WHH_S_ELEMS / 8; i += 256) {
    int lrow = i >> 7;            // 128 chunks (of 8 bf16) per row
    int kk   = (i & 127) << 3;
    int g = lrow >> 5, c = lrow & 31;
    *(uint4*)(whh_s + (size_t)lrow * EMB + kk) =
        *(const uint4*)(Whh + ((size_t)(g << 10) + ec0 + c) * EMB + kk);
  }
  // zero h_s rows 8..15 (padding rows, never rewritten)
  uint4 zz = make_uint4(0u, 0u, 0u, 0u);
  for (int i = threadIdx.x; i < (8 * EMB) / 8; i += 256)
    *(uint4*)(h_s + 8 * EMB + (i << 3)) = zz;
  lds_c[wave][lane] = 0.0f;
  __syncthreads();

  for (int t = 0; t < SEQ; ++t) {
    // stage h(t) (8 x 1024 bf16) global -> LDS
    const unsigned short* hin = hbuf + ((t & 1) << 13);
    for (int i = threadIdx.x; i < (8 * EMB) / 8; i += 256)
      *(uint4*)(h_s + (i << 3)) = *(const uint4*)(hin + (i << 3));
    __syncthreads();

    // acc = h @ Whh_slice^T entirely from LDS (ds_load_b128 + v_wmma)
    v8f acc = zero_acc();
#pragma unroll 4
    for (int k0 = 0; k0 < EMB; k0 += 32) {
      v16bf a = load_a_frag(h_s, EMB, 0, k0, lane);
      v16bf b = load_b_frag(whh_s, EMB, lrow0, k0, lane);
      acc = wmma_bf16(a, b, acc);
    }

    // scatter gate tile into LDS, fused with gates_x add (batch rows 0..7 only)
    {
      int col   = ((wave & 1) << 4) + (lane & 15);
      int mbase = (lane >> 4) << 3;
#pragma unroll
      for (int r = 0; r < 8; ++r) {
        int m = mbase + r;
        if (m < BB)
          lds_g[gate][m][col] =
              acc[r] + gates_x[((size_t)t * BB + m) * G4E + (gate << 10) + ec0 + col];
      }
    }
    __syncthreads();

    // c/h update: thread -> (batch = wave, e = lane)
    {
      int b = wave, e = lane;
      float gi = sigmoidf_(lds_g[0][b][e]);
      float gf = sigmoidf_(lds_g[1][b][e]);
      float gg = tanhf(lds_g[2][b][e]);
      float go = sigmoidf_(lds_g[3][b][e]);
      float c  = gf * lds_c[b][e] + gi * gg;
      lds_c[b][e] = c;
      float h = go * tanhf(c);
      rnn_out[((size_t)t * BB + b) * EMB + ec0 + e] = h;
      hbuf[(((t + 1) & 1) << 13) + b * EMB + ec0 + e] = f2bf(h);
    }
    __threadfence();
    __syncthreads();
    if (threadIdx.x == 0) {
      atomicAdd(syncv, 1);
      int target = (t + 1) * nblocks;
      while (atomicAdd(syncv, 0) < target) __builtin_amdgcn_s_sleep(1);
    }
    __syncthreads();
  }
}

// ---------------------------------------------------------------------------
// LayerNorm over E=1024, one block (256 thr) per row. Optional f32 output and
// up to two scaled bf16 outputs. Output layout mode: 0=[S,B,E], 1=[B,S,E],
// 2=[B,E,S] (pre-transposed for contiguous NT attention-mix GEMM).
// ---------------------------------------------------------------------------
__device__ inline size_t out_index(int mode, int row, int s, int b, int e) {
  if (mode == 1) return ((size_t)b * SEQ + s) * EMB + e;
  if (mode == 2) return ((size_t)b * EMB + e) * SEQ + s;
  return (size_t)row * EMB + e;
}

__global__ void __launch_bounds__(256) ln_dual_kernel(
    const float* __restrict__ in, const float* __restrict__ g,
    const float* __restrict__ bvec, float* __restrict__ outf,
    unsigned short* __restrict__ ob1, const float* __restrict__ sc1, int tr1,
    unsigned short* __restrict__ ob2, const float* __restrict__ sc2, int tr2) {
  __shared__ float red[256];
  int row = blockIdx.x;
  int tid = threadIdx.x;
  const float* x = in + (size_t)row * EMB;
  float vals[4];
  float lsum = 0.0f;
#pragma unroll
  for (int j = 0; j < 4; ++j) { vals[j] = x[tid + 256 * j]; lsum += vals[j]; }
  red[tid] = lsum; __syncthreads();
  for (int o = 128; o > 0; o >>= 1) { if (tid < o) red[tid] += red[tid + o]; __syncthreads(); }
  float mean = red[0] * (1.0f / EMB);
  __syncthreads();
  float lsq = 0.0f;
#pragma unroll
  for (int j = 0; j < 4; ++j) { float d = vals[j] - mean; lsq += d * d; }
  red[tid] = lsq; __syncthreads();
  for (int o = 128; o > 0; o >>= 1) { if (tid < o) red[tid] += red[tid + o]; __syncthreads(); }
  float rstd = rsqrtf(red[0] * (1.0f / EMB) + 1e-12f);
  int s = row >> 3, b = row & 7;  // row = s*B + b
#pragma unroll
  for (int j = 0; j < 4; ++j) {
    int e = tid + 256 * j;
    float y = (vals[j] - mean) * rstd * g[e] + bvec[e];
    if (outf) outf[(size_t)row * EMB + e] = y;
    if (ob1) ob1[out_index(tr1, row, s, b, e)] = f2bf(y * (sc1 ? sc1[e] : 1.0f));
    if (ob2) ob2[out_index(tr2, row, s, b, e)] = f2bf(y * (sc2 ? sc2[e] : 1.0f));
  }
}

// Softmax over K=S with causal mask; row (b,q) per block; writes bf16 weights.
__global__ void __launch_bounds__(256) softmax_kernel(
    const float* __restrict__ scores, const float* __restrict__ mask,
    unsigned short* __restrict__ w) {
  __shared__ float red[256];
  int bq = blockIdx.x, tid = threadIdx.x;
  int b = bq >> 11, q = bq & (SEQ - 1);
  const float* srow = scores + ((size_t)b * SEQ + q) * SEQ;
  const float* mrow = mask + (size_t)q * SEQ;
  float vals[8];
  float mx = -1e30f;
#pragma unroll
  for (int j = 0; j < 8; ++j) {
    int k = tid + 256 * j;
    vals[j] = srow[k] * 0.03125f + mrow[k];  // 1/sqrt(1024)
    mx = fmaxf(mx, vals[j]);
  }
  red[tid] = mx; __syncthreads();
  for (int o = 128; o > 0; o >>= 1) { if (tid < o) red[tid] = fmaxf(red[tid], red[tid + o]); __syncthreads(); }
  mx = red[0]; __syncthreads();
  float ssum = 0.0f;
#pragma unroll
  for (int j = 0; j < 8; ++j) { vals[j] = __expf(vals[j] - mx); ssum += vals[j]; }
  red[tid] = ssum; __syncthreads();
  for (int o = 128; o > 0; o >>= 1) { if (tid < o) red[tid] += red[tid + o]; __syncthreads(); }
  float inv = 1.0f / red[0];
  unsigned short* wrow = w + ((size_t)b * SEQ + q) * SEQ;
#pragma unroll
  for (int j = 0; j < 8; ++j) wrow[tid + 256 * j] = f2bf(vals[j] * inv);
}

// --- small elementwise / prep kernels --------------------------------------
__global__ void cvt_bf16_kernel(const float* __restrict__ in,
                                unsigned short* __restrict__ out, size_t n) {
  size_t i = (size_t)blockIdx.x * 256 + threadIdx.x;
  if (i < n) out[i] = f2bf(in[i]);
}
__global__ void bias_sum_kernel(const float* a, const float* b, float* o, int n) {
  int i = blockIdx.x * 256 + threadIdx.x;
  if (i < n) o[i] = a[i] + b[i];
}
// outs: [0..E)=qs_g, [E..2E)=ks_g, [2E..3E)=vs_g (overparam gate)
__global__ void scales_kernel(const float* qs, const float* ksp, const float* vsp,
                              const float* opW, const float* opb, float* outs) {
  int e = blockIdx.x * 256 + threadIdx.x;
  if (e >= EMB) return;
  outs[e]       = sigmoidf_(qs[e]);
  outs[EMB + e] = sigmoidf_(ksp[e]);
  float c = opb[e], f = opb[EMB + e];
  for (int k = 0; k < EMB; ++k) {
    float vo = sigmoidf_(vsp[k]);
    c += vo * opW[(size_t)e * EMB + k];
    f += vo * opW[(size_t)(EMB + e) * EMB + k];
  }
  outs[2 * EMB + e] = sigmoidf_(f) * tanhf(c);
}
__global__ void add_inplace_kernel(float* __restrict__ a,
                                   const float* __restrict__ b, size_t n) {
  size_t i = (size_t)blockIdx.x * 256 + threadIdx.x;
  if (i < n) a[i] += b[i];
}
__global__ void final_gelu_add_kernel(const float* __restrict__ hb,
                                      const float* __restrict__ lnff,
                                      float* __restrict__ out, size_t n) {
  size_t i = (size_t)blockIdx.x * 256 + threadIdx.x;
  if (i < n) {
    float x = hb[i];
    out[i] = x * sigmoidf_(1.702f * x) + lnff[i];
  }
}
__global__ void init_kernel(int* syncv, unsigned short* hbuf, int nh) {
  int i = blockIdx.x * 256 + threadIdx.x;
  if (i == 0) *syncv = 0;
  if (i < nh) hbuf[i] = 0;
}

// ---------------------------------------------------------------------------
extern "C" void kernel_launch(void* const* d_in, const int* in_sizes, int n_in,
                              void* d_out, int out_size, void* d_ws, size_t ws_size,
                              hipStream_t stream) {
  (void)in_sizes; (void)n_in; (void)out_size; (void)ws_size;
  const float* inputs    = (const float*)d_in[0];
  const float* attn_mask = (const float*)d_in[1];
  const float* Wih  = (const float*)d_in[3];
  const float* Whh  = (const float*)d_in[4];
  const float* bih  = (const float*)d_in[5];
  const float* bhh  = (const float*)d_in[6];
  const float* lnstart_g = (const float*)d_in[7],  *lnstart_b = (const float*)d_in[8];
  const float* lnmem_g   = (const float*)d_in[9],  *lnmem_b   = (const float*)d_in[10];
  const float* lnmid_g   = (const float*)d_in[11], *lnmid_b   = (const float*)d_in[12];
  const float* lnff_g    = (const float*)d_in[13], *lnff_b    = (const float*)d_in[14];
  const float* lnxff_g   = (const float*)d_in[15], *lnxff_b   = (const float*)d_in[16];
  const float* qln_g     = (const float*)d_in[17], *qln_b     = (const float*)d_in[18];
  const float* qs   = (const float*)d_in[19];
  const float* ksp  = (const float*)d_in[20];
  const float* vsp  = (const float*)d_in[21];
  const float* opW  = (const float*)d_in[22], *opb = (const float*)d_in[23];
  const float* qlW  = (const float*)d_in[24], *qlb = (const float*)d_in[25];
  const float* boomW = (const float*)d_in[26], *boomb = (const float*)d_in[27];
  float* out = (float*)d_out;

  // --- workspace carve-up (256B aligned) ---
  char* base = (char*)d_ws;
  size_t off = 0;
  auto alloc = [&](size_t bytes) -> char* {
    char* p = base + off;
    off += (bytes + 255) & ~(size_t)255;
    return p;
  };
  int*            syncv    = (int*)alloc(256);
  unsigned short* xln_bf   = (unsigned short*)alloc((size_t)SB * EMB * 2);
  unsigned short* Wih_bf   = (unsigned short*)alloc((size_t)G4E * EMB * 2);
  unsigned short* Whh_bf   = (unsigned short*)alloc((size_t)G4E * EMB * 2);
  unsigned short* qlW_bf   = (unsigned short*)alloc((size_t)EMB * EMB * 2);
  unsigned short* boomW_bf = (unsigned short*)alloc((size_t)EMB * EMB * 2);
  float*          bias_g   = (float*)alloc((size_t)G4E * 4);
  float*          scl      = (float*)alloc((size_t)3 * EMB * 4);   // qs_g|ks_g|vs_g
  float*          gates_x  = (float*)alloc((size_t)SB * G4E * 4);  // 256MB (reused)
  unsigned short* hbuf     = (unsigned short*)alloc((size_t)2 * BB * EMB * 2);
  float*          rnn_out  = (float*)alloc((size_t)SB * EMB * 4);  // reused as lnff
  unsigned short* k_bf     = (unsigned short*)alloc((size_t)BB * SEQ * EMB * 2);
  unsigned short* v_bf     = (unsigned short*)alloc((size_t)BB * SEQ * EMB * 2); // [B,E,S]
  float*          hmid     = (float*)alloc((size_t)SB * EMB * 4);
  unsigned short* hmid_bf  = (unsigned short*)alloc((size_t)SB * EMB * 2);
  float*          qtmp     = (float*)alloc((size_t)SB * EMB * 4);  // reused as hb
  unsigned short* q_bf     = (unsigned short*)alloc((size_t)BB * SEQ * EMB * 2);
  float*          h2       = (float*)alloc((size_t)SB * EMB * 4);
  unsigned short* xff_bf   = (unsigned short*)alloc((size_t)SB * EMB * 2);
  // overlays (regions dead by the time these are live):
  float*          scores   = gates_x;                                   // 128MB
  unsigned short* wsm      = (unsigned short*)((char*)gates_x + (size_t)BB * SEQ * SEQ * 4);
  float*          lnff_buf = rnn_out;
  float*          hb       = qtmp;

  const size_t nSBE = (size_t)SB * EMB;  // 16.7M

  // 0) init barrier counter + h0 = 0
  init_kernel<<<64, 256, 0, stream>>>(syncv, hbuf, 2 * BB * EMB);

  // 1) weight conversions + bias sum + attention gate scales
  cvt_bf16_kernel<<<(G4E * EMB + 255) / 256, 256, 0, stream>>>(Wih, Wih_bf, (size_t)G4E * EMB);
  cvt_bf16_kernel<<<(G4E * EMB + 255) / 256, 256, 0, stream>>>(Whh, Whh_bf, (size_t)G4E * EMB);
  cvt_bf16_kernel<<<(EMB * EMB + 255) / 256, 256, 0, stream>>>(qlW, qlW_bf, (size_t)EMB * EMB);
  cvt_bf16_kernel<<<(EMB * EMB + 255) / 256, 256, 0, stream>>>(boomW, boomW_bf, (size_t)EMB * EMB);
  bias_sum_kernel<<<16, 256, 0, stream>>>(bih, bhh, bias_g, G4E);
  scales_kernel<<<4, 256, 0, stream>>>(qs, ksp, vsp, opW, opb, scl);

  // 2) x_ln (bf16)
  ln_dual_kernel<<<SB, 256, 0, stream>>>(inputs, lnstart_g, lnstart_b, nullptr,
                                         xln_bf, nullptr, 0, nullptr, nullptr, 0);

  // 3) gates_x = x_ln @ Wih^T + (bih+bhh)   [16384 x 4096]
  gemm_bf16_kernel<<<dim3(16, 512, 1), 256, 0, stream>>>(
      xln_bf, 0, EMB, Wih_bf, 0, EMB, gates_x, 0, G4E, bias_g, SB, G4E, EMB);

  // 4) LSTM recurrence (persistent, 32 WGs, 288KB dynamic LDS, grid barrier/step)
  lstm_kernel<<<32, 256, (WHH_S_ELEMS + H_S_ELEMS) * sizeof(unsigned short), stream>>>(
      gates_x, Whh_bf, hbuf, rnn_out, syncv);

  // 5) mh = LN_mem(rnn_out) -> k = ks_g*mh [B,S,E], v = vs_g*mh [B,E,S]
  ln_dual_kernel<<<SB, 256, 0, stream>>>(rnn_out, lnmem_g, lnmem_b, nullptr,
                                         k_bf, scl + EMB, 1, v_bf, scl + 2 * EMB, 2);
  //    h_mid = LN_mid(rnn_out)  (f32 + bf16)
  ln_dual_kernel<<<SB, 256, 0, stream>>>(rnn_out, lnmid_g, lnmid_b, hmid,
                                         hmid_bf, nullptr, 0, nullptr, nullptr, 0);

  // 6) qtmp = h_mid @ qlW^T + qlb ; q = qs_g * LN_q(qtmp)  ([B,S,E] bf16)
  gemm_bf16_kernel<<<dim3(4, 512, 1), 256, 0, stream>>>(
      hmid_bf, 0, EMB, qlW_bf, 0, EMB, qtmp, 0, EMB, qlb, SB, EMB, EMB);
  ln_dual_kernel<<<SB, 256, 0, stream>>>(qtmp, qln_g, qln_b, nullptr,
                                         q_bf, scl, 1, nullptr, nullptr, 0);

  // 7) scores[b] = q[b] @ k[b]^T   (batched NT), softmax(+mask) -> w (bf16)
  gemm_bf16_kernel<<<dim3(8, 64, 8), 256, 0, stream>>>(
      q_bf, (long long)SEQ * EMB, EMB, k_bf, (long long)SEQ * EMB, EMB,
      scores, (long long)SEQ * SEQ, SEQ, nullptr, SEQ, SEQ, EMB);
  softmax_kernel<<<BB * SEQ, 256, 0, stream>>>(scores, attn_mask, wsm);

  // 8) mix[b] = w[b] @ v[b]; v pre-transposed [B,E,S] -> contiguous NT GEMM,
  //    written straight into [S,B,E] (ldc = B*E, batch offset = b*E)
  gemm_bf16_kernel<<<dim3(4, 64, 8), 256, 0, stream>>>(
      wsm, (long long)SEQ * SEQ, SEQ, v_bf, (long long)EMB * SEQ, SEQ,
      h2, (long long)EMB, BB * EMB, nullptr, SEQ, EMB, SEQ);
  add_inplace_kernel<<<(int)((nSBE + 255) / 256), 256, 0, stream>>>(h2, hmid, nSBE);

  // 9) Boom: out = gelu(LN_xff(h2) @ boomW^T + b) + LN_ff(h2)
  ln_dual_kernel<<<SB, 256, 0, stream>>>(h2, lnxff_g, lnxff_b, nullptr,
                                         xff_bf, nullptr, 0, nullptr, nullptr, 0);
  ln_dual_kernel<<<SB, 256, 0, stream>>>(h2, lnff_g, lnff_b, lnff_buf,
                                         nullptr, nullptr, 0, nullptr, nullptr, 0);
  gemm_bf16_kernel<<<dim3(4, 512, 1), 256, 0, stream>>>(
      xff_bf, 0, EMB, boomW_bf, 0, EMB, hb, 0, EMB, boomb, SB, EMB, EMB);
  final_gelu_add_kernel<<<(int)((nSBE + 255) / 256), 256, 0, stream>>>(hb, lnff_buf, out, nSBE);
}
